// DynamicWeightConv_56100862820380
// MI455X (gfx1250) — matrix operand order repaired and checked by reading, not compiled, not run on hardware
//
#include <hip/hip_runtime.h>
#include <hip/hip_bf16.h>
#include <stdint.h>

// ---------------------------------------------------------------------------
// DynamicWeightConv on gfx1250 (MI455X):
//   gap -> mlp1 -> mlp2 -> kgen (WMMA bf16 GEMM, emits bf16 kernels)
//       -> [xcvt: x fp32 -> bf16 once]
//       -> conv  (implicit-GEMM per batch, v_wmma_f32_16x16x32_bf16,
//                 async global->LDS (ASYNCcnt) for the kernel slab)
// ---------------------------------------------------------------------------

typedef __attribute__((ext_vector_type(16))) __bf16 v16bf;
typedef __attribute__((ext_vector_type(8)))  float  v8f;

typedef int v4i __attribute__((vector_size(16)));
typedef __attribute__((address_space(1))) v4i g4v;   // global 16B vector
typedef __attribute__((address_space(3))) v4i l4v;   // LDS 16B vector

#if defined(__has_builtin)
#if __has_builtin(__builtin_amdgcn_global_load_async_to_lds_b128) && \
    __has_builtin(__builtin_amdgcn_s_wait_asynccnt)
#define ASYNC_LDS 1
#endif
#endif
#ifndef ASYNC_LDS
#define ASYNC_LDS 0
#endif

#define BATCH 32
#define CIN   64
#define HH    64
#define WW    64
#define OUTC  128
#define KTOT  576        // CIN * 3 * 3
#define LAT   64
#define JTOT  73728      // OUTC * KTOT

// fp32 -> bf16, round-to-nearest-even
__device__ __forceinline__ unsigned short f2bf(float f) {
  union { float f; unsigned int u; } v; v.f = f;
  unsigned int u = v.u;
  u += 0x7FFFu + ((u >> 16) & 1u);
  return (unsigned short)(u >> 16);
}

// assemble a 16-half WMMA fragment from two 16B LDS chunks (ds_load_b128 x2)
__device__ __forceinline__ v16bf frag_from(const unsigned short* p0,
                                           const unsigned short* p1) {
  union { uint4 q[2]; v16bf v; } u;
  u.q[0] = *(const uint4*)p0;
  u.q[1] = *(const uint4*)p1;
  return u.v;
}

// ------------------------------ GAP ---------------------------------------
__global__ void __launch_bounds__(256)
gap_kernel(const float* __restrict__ x, float* __restrict__ gap) {
  __shared__ float red[256];
  int bc = blockIdx.x;                       // (b*C + c), 0..2047
  const float* p = x + (size_t)bc * (HH * WW);
  float s = 0.f;
  for (int i = threadIdx.x; i < HH * WW; i += 256) s += p[i];
  red[threadIdx.x] = s;
  __syncthreads();
  for (int off = 128; off > 0; off >>= 1) {
    if ((int)threadIdx.x < off) red[threadIdx.x] += red[threadIdx.x + off];
    __syncthreads();
  }
  if (threadIdx.x == 0) gap[bc] = red[0] * (1.0f / (HH * WW));
}

// ------------------------------ MLP layer ----------------------------------
__global__ void __launch_bounds__(256)
mlp_kernel(const float* __restrict__ in, const float* __restrict__ Wm,
           const float* __restrict__ bm, float* __restrict__ outp) {
  int t = blockIdx.x * 256 + threadIdx.x;    // 0..2047
  int b = t >> 6, j = t & 63;
  const float* row = in + b * LAT;
  float acc = bm[j];
#pragma unroll 8
  for (int k = 0; k < LAT; ++k) acc += row[k] * Wm[k * LAT + j];
  outp[t] = fmaxf(acc, 0.f);
}

// ------------------------- x fp32 -> bf16 (once) ---------------------------
__global__ void __launch_bounds__(256)
xcvt_kernel(const float* __restrict__ x, unsigned short* __restrict__ xbf) {
  int i = (blockIdx.x * 256 + threadIdx.x) * 4;
  float4 v = *(const float4*)(x + i);
  ushort4 o;
  o.x = f2bf(v.x); o.y = f2bf(v.y); o.z = f2bf(v.z); o.w = f2bf(v.w);
  *(ushort4*)(xbf + i) = o;
}

// --------------------- kernel generator GEMM (WMMA) ------------------------
// ker[b][j] = sum_k f2[b][k] * Wf[k][j] + bf[j]   -> bf16
__global__ void __launch_bounds__(256)
kgen_kernel(const float* __restrict__ f2, const float* __restrict__ Wf,
            const float* __restrict__ bfv, unsigned short* __restrict__ kerb) {
  __shared__ unsigned short f2s[BATCH * LAT];   // [32][64] bf16, M-major
  __shared__ unsigned short wt[128 * 40];       // [n][k] bf16, N-major (padded)
  int t = threadIdx.x;
  int jt = blockIdx.x * 128;

  for (int i = 0; i < 8; ++i) {                 // stage f2 as bf16
    int idx = i * 256 + t;
    f2s[idx] = f2bf(f2[idx]);
  }

  int lane = t & 31, wv = t >> 5;
  int col = lane & 15;
  int kb  = (lane < 16) ? 0 : 8;                // A-frag K base
  int kb2 = (lane < 16) ? 0 : 16;               // unused; B-frag uses kb
  int n0  = wv * 16;

  v8f c0 = {}, c1 = {};
  for (int k0 = 0; k0 < LAT; k0 += 32) {
    __syncthreads();
#pragma unroll
    for (int p = 0; p < 16; ++p) {
      int idx = p * 256 + t;                    // 0..4095
      int kkL = idx >> 7, n = idx & 127;
      wt[n * 40 + kkL] = f2bf(Wf[(size_t)(k0 + kkL) * JTOT + jt + n]);
    }
    __syncthreads();
    v16bf bfr = frag_from(&wt[(n0 + col) * 40 + kb],
                          &wt[(n0 + col) * 40 + kb + 16]);
    v16bf a0  = frag_from(&f2s[col * LAT + k0 + kb],
                          &f2s[col * LAT + k0 + kb + 16]);
    v16bf a1  = frag_from(&f2s[(16 + col) * LAT + k0 + kb],
                          &f2s[(16 + col) * LAT + k0 + kb + 16]);
    c0 = __builtin_amdgcn_wmma_f32_16x16x32_bf16(false, a0, false, bfr,
                                                 (short)0, c0, false, false);
    c1 = __builtin_amdgcn_wmma_f32_16x16x32_bf16(false, a1, false, bfr,
                                                 (short)0, c1, false, false);
  }

  int j = jt + n0 + col;
  float bias = bfv[j];
  int rb = (lane < 16) ? 0 : 8;
#pragma unroll
  for (int r = 0; r < 8; ++r) {
    kerb[(size_t)(r + rb) * JTOT + j]      = f2bf(c0[r] + bias);
    kerb[(size_t)(16 + r + rb) * JTOT + j] = f2bf(c1[r] + bias);
  }
}

// --------------------- grouped dynamic conv (WMMA) -------------------------
// Implicit GEMM per batch: M=O=128, N=W row (64), K=576 in chunks of 32.
template <bool XBF>
__global__ void __launch_bounds__(256)
conv_kernel(const float* __restrict__ x, const unsigned short* __restrict__ xbf,
            const unsigned short* __restrict__ kerb, float* __restrict__ out) {
  __shared__ unsigned short At[OUTC * 40];   // [o][k] bf16 (padded rows)
  __shared__ unsigned short Bt[WW * 40];     // [w][k] bf16 (im2col transposed)
  int t = threadIdx.x;
  int h = blockIdx.x;                        // image row
  int b = blockIdx.y;                        // batch
  int lane = t & 31, wv = t >> 5;
  int col = lane & 15;
  int kb  = (lane < 16) ? 0 : 8;
  int kb2 = (lane < 16) ? 0 : 16;

  const unsigned short* kerRow = kerb + (size_t)b * JTOT;
  const float*          xb  = x   + (size_t)b * (CIN * HH * WW);
  const unsigned short* xbb = XBF ? xbf + (size_t)b * (CIN * HH * WW) : nullptr;

  v8f acc[4] = {};
  for (int k0 = 0; k0 < KTOT; k0 += 32) {
    __syncthreads();
    // A tile: 128 x 32 bf16 kernel slab
#pragma unroll
    for (int it = 0; it < 2; ++it) {
      int chunk = it * 256 + t;              // 0..511
      int o  = chunk >> 2;
      int ko = (chunk & 3) << 3;
      const unsigned short* gp = kerRow + (size_t)o * KTOT + k0 + ko;
      unsigned short* lp = &At[o * 40 + ko];
#if ASYNC_LDS
      __builtin_amdgcn_global_load_async_to_lds_b128(
          (g4v*)(uintptr_t)gp, (l4v*)lp, 0, 0);
#else
      *(uint4*)lp = *(const uint4*)gp;
#endif
    }
    // B tile: on-the-fly im2col (zero-padded)
#pragma unroll
    for (int p = 0; p < 8; ++p) {
      int idx = p * 256 + t;                 // 0..2047
      int kkL = idx >> 6, n = idx & 63;
      int kk = k0 + kkL;
      int c  = kk / 9;
      int r  = kk - 9 * c;
      int kh = r / 3;
      int kw = r - 3 * kh;
      int h2 = h + kh - 1;
      int w2 = n + kw - 1;
      bool inb = (unsigned)h2 < (unsigned)HH && (unsigned)w2 < (unsigned)WW;
      unsigned short val;
      if (XBF) {
        val = inb ? xbb[((c << 6) + h2) * WW + w2] : (unsigned short)0;
      } else {
        val = f2bf(inb ? xb[((c << 6) + h2) * WW + w2] : 0.f);
      }
      Bt[n * 40 + kkL] = val;
    }
    if (k0 + 32 < KTOT)  // pull next kernel slab toward L2
      __builtin_prefetch(kerRow + (size_t)(wv * 16) * KTOT + k0 + 32, 0, 1);
#if ASYNC_LDS
    __builtin_amdgcn_s_wait_asynccnt(0);
#endif
    __syncthreads();

    v16bf a = frag_from(&At[(wv * 16 + col) * 40 + kb],
                        &At[(wv * 16 + col) * 40 + kb + 16]);
#pragma unroll
    for (int j = 0; j < 4; ++j) {
      v16bf bfrag = frag_from(&Bt[(j * 16 + col) * 40 + kb],
                              &Bt[(j * 16 + col) * 40 + kb + 16]);
      acc[j] = __builtin_amdgcn_wmma_f32_16x16x32_bf16(false, a, false, bfrag,
                                                       (short)0, acc[j],
                                                       false, false);
    }
  }

  int rb = (lane < 16) ? 0 : 8;
#pragma unroll
  for (int j = 0; j < 4; ++j) {
    int w = j * 16 + col;
    float* op = out + (((size_t)b * OUTC + wv * 16 + rb) * HH + h) * WW + w;
#pragma unroll
    for (int r = 0; r < 8; ++r)
      op[(size_t)r * (HH * WW)] = acc[j][r];
  }
}

// ---------------------------------------------------------------------------
extern "C" void kernel_launch(void* const* d_in, const int* in_sizes, int n_in,
                              void* d_out, int out_size, void* d_ws,
                              size_t ws_size, hipStream_t stream) {
  const float* x   = (const float*)d_in[0];
  const float* W1  = (const float*)d_in[1];
  const float* b1  = (const float*)d_in[2];
  const float* W2  = (const float*)d_in[3];
  const float* b2  = (const float*)d_in[4];
  const float* Wf  = (const float*)d_in[5];
  const float* bfv = (const float*)d_in[6];
  float* out = (float*)d_out;

  const size_t kerbBytes = (size_t)BATCH * JTOT * 2;            // 4.5 MiB
  const size_t xbfBytes  = (size_t)BATCH * CIN * HH * WW * 2;   // 16 MiB
  const size_t smallBytes = 3u * BATCH * LAT * 4;

  char* ws = (char*)d_ws;
  unsigned short* kerb = (unsigned short*)ws;
  size_t off = kerbBytes;

  bool use_xbf = ws_size >= kerbBytes + xbfBytes + smallBytes;
  unsigned short* xbf = nullptr;
  if (use_xbf) { xbf = (unsigned short*)(ws + off); off += xbfBytes; }

  float* gap = (float*)(ws + off);
  float* f1  = gap + BATCH * LAT;
  float* f2  = f1  + BATCH * LAT;

  gap_kernel<<<BATCH * CIN, 256, 0, stream>>>(x, gap);
  mlp_kernel<<<(BATCH * LAT) / 256, 256, 0, stream>>>(gap, W1, b1, f1);
  mlp_kernel<<<(BATCH * LAT) / 256, 256, 0, stream>>>(f1, W2, b2, f2);
  kgen_kernel<<<JTOT / 128, 256, 0, stream>>>(f2, Wf, bfv, kerb);
  if (use_xbf)
    xcvt_kernel<<<(BATCH * CIN * HH * WW) / 1024, 256, 0, stream>>>(x, xbf);

  dim3 cg(HH, BATCH);
  if (use_xbf)
    conv_kernel<true><<<cg, 256, 0, stream>>>(x, xbf, kerb, out);
  else
    conv_kernel<false><<<cg, 256, 0, stream>>>(x, nullptr, kerb, out);
}
